// ChytorchDiscrete_29033978921322
// MI455X (gfx1250) — matrix operand order repaired and checked by reading, not compiled, not run on hardware
//
#include <hip/hip_runtime.h>
#include <hip/hip_bf16.h>
#include <math.h>

typedef __bf16 bf16;
typedef __attribute__((ext_vector_type(16))) __bf16 v16bf;
typedef __attribute__((ext_vector_type(8)))  __bf16 v8bf;
typedef __attribute__((ext_vector_type(8)))  float  v8f;

#define B_GRAPHS 64
#define N_ATOMS  127
#define SEQ      128
#define DM       1024
#define NH       16
#define HD       64
#define DFF      3072
#define TOK      (B_GRAPHS * SEQ)   // 8192 tokens

// ---------------------------------------------------------------------------
// CDNA5 primitives
// ---------------------------------------------------------------------------
static __device__ __forceinline__ v8f wmma_bf16(v16bf a, v16bf b, v8f c) {
  // (neg_a, A, neg_b, B, c_mod, C, reuse_a, reuse_b)
  return __builtin_amdgcn_wmma_f32_16x16x32_bf16(false, a, false, b, (short)0, c,
                                                 false, false);
}

// Async global -> LDS copy of 16 bytes per lane (GLOBAL_LOAD_ASYNC_TO_LDS_B128,
// tracked by ASYNCcnt). lds_off is the LDS byte offset = low 32 bits of the
// flat address of a __shared__ pointer (ISA 10.2).
static __device__ __forceinline__ void async_g2l_b128(unsigned int lds_off,
                                                      const bf16* gsrc) {
  asm volatile("global_load_async_to_lds_b128 %0, %1, off"
               :: "v"(lds_off), "v"(gsrc)
               : "memory");
}

static __device__ __forceinline__ void wait_async0() {
  asm volatile("s_wait_asynccnt 0x0" ::: "memory");
}

static __device__ __forceinline__ unsigned int lds_off_of(const void* p) {
  return (unsigned int)(uintptr_t)p;
}

// A fragment: A is row-major (M x K, leading dim lda). 16x32 tile at (m0,k0).
// Lanes 0-15 -> M=0..15 with K {0..7,16..23}; lanes 16-31 -> K {8..15,24..31}.
static __device__ __forceinline__ v16bf load_a_frag(const bf16* __restrict__ A,
                                                    int lda, int m0, int k0, int lane) {
  int row = m0 + (lane & 15);
  int kb  = k0 + ((lane >> 4) << 3);           // +0 or +8
  const bf16* p = A + (size_t)row * lda + kb;
  v8bf lo = *(const v8bf*)p;                   // K = kb .. kb+7
  v8bf hi = *(const v8bf*)(p + 16);            // K = kb+16 .. kb+23
  return __builtin_shufflevector(lo, hi, 0,1,2,3,4,5,6,7,8,9,10,11,12,13,14,15);
}

// B fragment: B is row-major (K x N, leading dim ldb). 32x16 tile at (k0,n0).
// lane -> N column (n0+lane%16); lanes 0-15 K {0..7,16..23}, lanes 16-31 K {8..15,24..31}.
static __device__ __forceinline__ v16bf load_b_frag(const bf16* __restrict__ Bm,
                                                    int ldb, int k0, int n0, int lane) {
  const bf16* p = Bm + (size_t)(k0 + ((lane >> 4) << 3)) * ldb + n0 + (lane & 15);
  v16bf b;
#pragma unroll
  for (int i = 0; i < 8; ++i) {
    b[i]     = p[(size_t)i * ldb];
    b[8 + i] = p[(size_t)(16 + i) * ldb];
  }
  return b;
}

// ---------------------------------------------------------------------------
// Generic tiled WMMA GEMM:  C[M,N] = act(A[M,K] * W[K,N] + bias)
// 256 threads = 8 waves; block tile 128(M) x 128(N); wave tile 16 x 128.
// Weight tiles are 64(K) x 128(N) bf16, double-buffered in LDS via async
// global->LDS DMA (one barrier per 64 K, 16 wmma per stage per wave).
// act: 0 = none, 1 = exact GELU. Cf (f32) and Cb (bf16) outputs optional.
// ---------------------------------------------------------------------------
__global__ __launch_bounds__(256)
void gemm_wmma_kernel(const bf16* __restrict__ A, const bf16* __restrict__ W,
                      const float* __restrict__ bias, float* __restrict__ Cf,
                      bf16* __restrict__ Cb, int M, int N, int K, int act) {
  __shared__ __align__(32) bf16 sW[2][64 * 128];   // 2 x 16 KB
  const int lane = threadIdx.x & 31;
  const int wave = threadIdx.x >> 5;
  const int n0   = blockIdx.x * 128;
  const int m0   = blockIdx.y * 128 + wave * 16;

  v8f acc[8];
#pragma unroll
  for (int t = 0; t < 8; ++t)
#pragma unroll
    for (int e = 0; e < 8; ++e) acc[t][e] = 0.0f;

  const int r = threadIdx.x >> 2;          // 0..63 : K row of the LDS tile
  const int c = (threadIdx.x & 3) * 32;    // 0,32,64,96 : N column chunk (32 bf16)
  const bf16* wsrc = W + (size_t)r * N + n0 + c;                  // tile 0 source
  const unsigned int d0 = lds_off_of(&sW[0][r * 128 + c]);
  const unsigned int d1 = lds_off_of(&sW[1][r * 128 + c]);

  // prefetch tile 0 into buffer 0 (4 x 16B per thread)
#pragma unroll
  for (int u = 0; u < 4; ++u) async_g2l_b128(d0 + u * 16, wsrc + u * 8);

  const int nstages = K / 64;
  for (int s = 0; s < nstages; ++s) {
    const int kt = s * 64;
    const bf16* sbuf = &sW[s & 1][0];
    // drain our async writes, then block-wide barrier so every wave's tile
    // rows are visible before any wave reads the buffer.
    wait_async0();
    __syncthreads();
    if (s + 1 < nstages) {                 // prefetch next tile into other buffer
      const bf16* nsrc = wsrc + (size_t)(kt + 64) * N;
      const unsigned int nd = (s & 1) ? d0 : d1;
#pragma unroll
      for (int u = 0; u < 4; ++u) async_g2l_b128(nd + u * 16, nsrc + u * 8);
    }
#pragma unroll
    for (int ks = 0; ks < 64; ks += 32) {
      v16bf a = load_a_frag(A, K, m0, kt + ks, lane);
#pragma unroll
      for (int g = 0; g < 2; ++g) {        // grouped B preload -> batched wmma
        v16bf bfr[4];
#pragma unroll
        for (int t = 0; t < 4; ++t)
          bfr[t] = load_b_frag(sbuf, 128, ks, (g * 4 + t) * 16, lane);
#pragma unroll
        for (int t = 0; t < 4; ++t) acc[g * 4 + t] = wmma_bf16(a, bfr[t], acc[g * 4 + t]);
      }
    }
  }

  const int half = lane >> 4, nlo = lane & 15;
#pragma unroll
  for (int t = 0; t < 8; ++t) {
    int col = n0 + t * 16 + nlo;
    float bv = bias ? bias[col] : 0.0f;
#pragma unroll
    for (int e = 0; e < 8; ++e) {
      int row = m0 + e + 8 * half;
      float v = acc[t][e] + bv;
      if (act == 1) v = 0.5f * v * (1.0f + erff(v * 0.70710678118654752f));
      if (Cf) Cf[(size_t)row * N + col] = v;
      if (Cb) Cb[(size_t)row * N + col] = (bf16)v;
    }
  }
}

// ---------------------------------------------------------------------------
// QKV GEMM (M=TOK, N=3*DM, K=DM), same pipeline; epilogue scatters into
// attention layouts: q (B,NH,SEQ,HD), k transposed (B,NH,HD,SEQ), v (B,NH,SEQ,HD).
// ---------------------------------------------------------------------------
__global__ __launch_bounds__(256)
void gemm_qkv_kernel(const bf16* __restrict__ A, const bf16* __restrict__ W,
                     const float* __restrict__ bias, bf16* __restrict__ qo,
                     bf16* __restrict__ ko, bf16* __restrict__ vo) {
  __shared__ __align__(32) bf16 sW[2][64 * 128];
  const int N = 3 * DM, K = DM;
  const int lane = threadIdx.x & 31;
  const int wave = threadIdx.x >> 5;
  const int n0   = blockIdx.x * 128;
  const int m0   = blockIdx.y * 128 + wave * 16;

  v8f acc[8];
#pragma unroll
  for (int t = 0; t < 8; ++t)
#pragma unroll
    for (int e = 0; e < 8; ++e) acc[t][e] = 0.0f;

  const int r = threadIdx.x >> 2;
  const int c = (threadIdx.x & 3) * 32;
  const bf16* wsrc = W + (size_t)r * N + n0 + c;
  const unsigned int d0 = lds_off_of(&sW[0][r * 128 + c]);
  const unsigned int d1 = lds_off_of(&sW[1][r * 128 + c]);

#pragma unroll
  for (int u = 0; u < 4; ++u) async_g2l_b128(d0 + u * 16, wsrc + u * 8);

  const int nstages = K / 64;
  for (int s = 0; s < nstages; ++s) {
    const int kt = s * 64;
    const bf16* sbuf = &sW[s & 1][0];
    wait_async0();
    __syncthreads();
    if (s + 1 < nstages) {
      const bf16* nsrc = wsrc + (size_t)(kt + 64) * N;
      const unsigned int nd = (s & 1) ? d0 : d1;
#pragma unroll
      for (int u = 0; u < 4; ++u) async_g2l_b128(nd + u * 16, nsrc + u * 8);
    }
#pragma unroll
    for (int ks = 0; ks < 64; ks += 32) {
      v16bf a = load_a_frag(A, K, m0, kt + ks, lane);
#pragma unroll
      for (int g = 0; g < 2; ++g) {
        v16bf bfr[4];
#pragma unroll
        for (int t = 0; t < 4; ++t)
          bfr[t] = load_b_frag(sbuf, 128, ks, (g * 4 + t) * 16, lane);
#pragma unroll
        for (int t = 0; t < 4; ++t) acc[g * 4 + t] = wmma_bf16(a, bfr[t], acc[g * 4 + t]);
      }
    }
  }

  const int half = lane >> 4, nlo = lane & 15;
#pragma unroll
  for (int t = 0; t < 8; ++t) {
    int col  = n0 + t * 16 + nlo;          // 0..3071
    int part = col >> 10;                  // 0:q 1:k 2:v
    int w    = col & 1023;
    int h    = w >> 6;
    int hd   = w & 63;
    float bv = bias[col];
#pragma unroll
    for (int e = 0; e < 8; ++e) {
      int row = m0 + e + 8 * half;         // global token
      int bb  = row >> 7;
      int l   = row & 127;
      bf16 val = (bf16)(acc[t][e] + bv);
      if (part == 0)      qo[(((size_t)bb * NH + h) * SEQ + l) * HD + hd] = val;
      else if (part == 1) ko[(((size_t)bb * NH + h) * HD + hd) * SEQ + l] = val;
      else                vo[(((size_t)bb * NH + h) * SEQ + l) * HD + hd] = val;
    }
  }
}

// ---------------------------------------------------------------------------
// Fused attention per (b,h) plane: S = Q K^T * scale + mask, softmax, O = P V.
// 8 waves, wave w owns rows [16w,16w+16). P staged to LDS as bf16 to become
// the A-fragment of the second WMMA matmul. Writes O into (B,SEQ,DM) bf16.
// ---------------------------------------------------------------------------
__global__ __launch_bounds__(256)
void attn_wmma_kernel(const bf16* __restrict__ q, const bf16* __restrict__ kt,
                      const bf16* __restrict__ v, const float* __restrict__ mask,
                      bf16* __restrict__ out) {
  __shared__ __align__(32) bf16 sP[8 * 16 * 128];   // 32 KB
  const int plane = blockIdx.x;                     // b*NH + h
  const int bb = plane >> 4, h = plane & 15;
  const int lane = threadIdx.x & 31;
  const int wave = threadIdx.x >> 5;
  const int m0 = wave * 16;

  const bf16*  Q  = q    + (size_t)plane * SEQ * HD;
  const bf16*  Kt = kt   + (size_t)plane * HD * SEQ;
  const bf16*  V  = v    + (size_t)plane * SEQ * HD;
  const float* Mp = mask + (size_t)plane * SEQ * SEQ;

  // ---- S = Q K^T : 8 tiles of 16x16 across the 128 key columns ----
  v8f acc[8];
#pragma unroll
  for (int t = 0; t < 8; ++t)
#pragma unroll
    for (int e = 0; e < 8; ++e) acc[t][e] = 0.0f;

#pragma unroll
  for (int k0 = 0; k0 < HD; k0 += 32) {
    v16bf a = load_a_frag(Q, HD, m0, k0, lane);
#pragma unroll
    for (int g = 0; g < 2; ++g) {
      v16bf bfr[4];
#pragma unroll
      for (int t = 0; t < 4; ++t) bfr[t] = load_b_frag(Kt, SEQ, k0, (g * 4 + t) * 16, lane);
#pragma unroll
      for (int t = 0; t < 4; ++t) acc[g * 4 + t] = wmma_bf16(a, bfr[t], acc[g * 4 + t]);
    }
  }

  const int half = lane >> 4, nlo = lane & 15;
  // scale + distance-bias mask
#pragma unroll
  for (int t = 0; t < 8; ++t) {
    int col = t * 16 + nlo;
#pragma unroll
    for (int e = 0; e < 8; ++e) {
      int row = m0 + e + 8 * half;
      acc[t][e] = acc[t][e] * 0.125f + Mp[(size_t)row * SEQ + col];
    }
  }

  // softmax along 128 columns of each row (8 per-lane vals x 16 lanes/half)
  float inv[8];
#pragma unroll
  for (int e = 0; e < 8; ++e) {
    float m = -INFINITY;
#pragma unroll
    for (int t = 0; t < 8; ++t) m = fmaxf(m, acc[t][e]);
#pragma unroll
    for (int d = 1; d <= 8; d <<= 1) m = fmaxf(m, __shfl_xor(m, d, 32));
    float s = 0.0f;
#pragma unroll
    for (int t = 0; t < 8; ++t) { float p = __expf(acc[t][e] - m); acc[t][e] = p; s += p; }
#pragma unroll
    for (int d = 1; d <= 8; d <<= 1) s += __shfl_xor(s, d, 32);
    inv[e] = 1.0f / s;
  }

  // stage P (16x128 bf16 per wave) into LDS in row-major A layout
  bf16* Pw = sP + (size_t)wave * 16 * 128;
#pragma unroll
  for (int t = 0; t < 8; ++t) {
    int col = t * 16 + nlo;
#pragma unroll
    for (int e = 0; e < 8; ++e) {
      int rl = e + 8 * half;
      Pw[rl * 128 + col] = (bf16)(acc[t][e] * inv[e]);
    }
  }
  __syncthreads();

  // ---- O = P V : 4 tiles of 16x16 across the 64 head dims ----
  v8f oacc[4];
#pragma unroll
  for (int t = 0; t < 4; ++t)
#pragma unroll
    for (int e = 0; e < 8; ++e) oacc[t][e] = 0.0f;

#pragma unroll
  for (int k0 = 0; k0 < SEQ; k0 += 32) {
    v16bf a = load_a_frag(Pw, 128, 0, k0, lane);
    v16bf bfr[4];
#pragma unroll
    for (int t = 0; t < 4; ++t) bfr[t] = load_b_frag(V, HD, k0, t * 16, lane);
#pragma unroll
    for (int t = 0; t < 4; ++t) oacc[t] = wmma_bf16(a, bfr[t], oacc[t]);
  }

#pragma unroll
  for (int t = 0; t < 4; ++t) {
    int col = t * 16 + nlo;
#pragma unroll
    for (int e = 0; e < 8; ++e) {
      int row = m0 + e + 8 * half;     // token in plane
      out[((size_t)bb * SEQ + row) * DM + h * HD + col] = (bf16)oacc[t][e];
    }
  }
}

// ---------------------------------------------------------------------------
// Residual add + LayerNorm over D=1024, one block (256 thr) per token row.
// Writes both f32 (next residual) and bf16 (next GEMM A operand).
// ---------------------------------------------------------------------------
__global__ __launch_bounds__(256)
void add_ln_kernel(const float* __restrict__ xin, const float* __restrict__ yin,
                   const float* __restrict__ g, const float* __restrict__ bta,
                   float* __restrict__ xof, bf16* __restrict__ xob) {
  __shared__ float red[16];
  const int row = blockIdx.x;
  const float* xr = xin + (size_t)row * DM;
  const float* yr = yin + (size_t)row * DM;

  float vals[4], s = 0.0f, s2 = 0.0f;
#pragma unroll
  for (int i = 0; i < 4; ++i) {
    int c = threadIdx.x + i * 256;
    float t = xr[c] + yr[c];
    vals[i] = t; s += t; s2 += t * t;
  }
#pragma unroll
  for (int d = 1; d < 32; d <<= 1) { s += __shfl_xor(s, d, 32); s2 += __shfl_xor(s2, d, 32); }
  const int lane = threadIdx.x & 31, wave = threadIdx.x >> 5;
  if (lane == 0) { red[wave] = s; red[8 + wave] = s2; }
  __syncthreads();
  if (threadIdx.x == 0) {
    float a = 0.0f, b = 0.0f;
    for (int i = 0; i < 8; ++i) { a += red[i]; b += red[8 + i]; }
    float mean = a / (float)DM;
    float var  = b / (float)DM - mean * mean;
    red[0] = mean; red[1] = rsqrtf(var + 1e-5f);
  }
  __syncthreads();
  const float mean = red[0], rstd = red[1];
#pragma unroll
  for (int i = 0; i < 4; ++i) {
    int c = threadIdx.x + i * 256;
    float t = (vals[i] - mean) * rstd * g[c] + bta[c];
    xof[(size_t)row * DM + c] = t;
    xob[(size_t)row * DM + c] = (bf16)t;
  }
}

// ---------------------------------------------------------------------------
// One-time kernels
// ---------------------------------------------------------------------------
__global__ void cvt_f32_bf16_kernel(const float* __restrict__ src,
                                    bf16* __restrict__ dst, int n) {
  int i = blockIdx.x * blockDim.x + threadIdx.x;
  if (i < n) dst[i] = (bf16)src[i];
}

__global__ void embed_kernel(const int* __restrict__ z, const int* __restrict__ hgs,
                             const float* __restrict__ aemb, const float* __restrict__ nemb,
                             float* __restrict__ xf, bf16* __restrict__ xb) {
  int idx = blockIdx.x * blockDim.x + threadIdx.x;   // over TOK*DM
  if (idx >= TOK * DM) return;
  int d = idx & 1023, t = idx >> 10;
  int bb = t >> 7, l = t & 127;
  int ai, ni;
  if (l == 0) { ai = 1; ni = 0; }                     // cls token
  else { int a = bb * N_ATOMS + l - 1; ai = z[a] + 2; ni = hgs[a] + 2; }
  float vv = aemb[(size_t)ai * DM + d] + nemb[(size_t)ni * DM + d];
  xf[idx] = vv; xb[idx] = (bf16)vv;
}

__global__ void mask_kernel(const float* __restrict__ pos, const float* __restrict__ demb,
                            float* __restrict__ mask) {
  int idx = blockIdx.x * blockDim.x + threadIdx.x;   // over B*SEQ*SEQ
  if (idx >= B_GRAPHS * SEQ * SEQ) return;
  int j = idx & 127, rest = idx >> 7;
  int i = rest & 127, bb = rest >> 7;
  int didx;
  if (i == 0 || j == 0) {
    didx = 1;                                        // cls row/col
  } else {
    const float* pi = pos + ((size_t)bb * N_ATOMS + i - 1) * 3;
    const float* pj = pos + ((size_t)bb * N_ATOMS + j - 1) * 3;
    float dx = pi[0] - pj[0], dy = pi[1] - pj[1], dz = pi[2] - pj[2];
    float d = sqrtf(dx * dx + dy * dy + dz * dz);
    int cnt = 0;                                     // np.digitize = #(bins <= d)
#pragma unroll
    for (int t = 0; t < 85; ++t) {
      float bn = (t == 0) ? -1.0f : (t == 1) ? 0.0f : (t == 2) ? 0.01f
                                             : (0.75f + 0.05f * (float)t);
      cnt += (d >= bn) ? 1 : 0;
    }
    didx = cnt;
  }
#pragma unroll
  for (int h = 0; h < NH; ++h) {
    float vv = (didx == 0) ? -INFINITY : demb[didx * NH + h];
    mask[(((size_t)bb * NH + h) * SEQ + i) * SEQ + j] = vv;
  }
}

__global__ void cls_extract_kernel(const float* __restrict__ x, float* __restrict__ out) {
  int i = blockIdx.x * blockDim.x + threadIdx.x;     // over B*DM
  if (i < B_GRAPHS * DM) {
    int bb = i >> 10, d = i & 1023;
    out[i] = x[((size_t)bb * SEQ) * DM + d];
  }
}

// ---------------------------------------------------------------------------
// Host launcher
// ---------------------------------------------------------------------------
extern "C" void kernel_launch(void* const* d_in, const int* in_sizes, int n_in,
                              void* d_out, int out_size, void* d_ws, size_t ws_size,
                              hipStream_t stream) {
  const int*   z    = (const int*)d_in[0];
  const int*   hgs  = (const int*)d_in[1];
  const float* pos  = (const float*)d_in[2];
  // d_in[3] = batch (uniform, unused)
  const float* aemb = (const float*)d_in[4];
  const float* nemb = (const float*)d_in[5];
  const float* demb = (const float*)d_in[6];
  const float* Wqkv = (const float*)d_in[7];
  const float* bqkv = (const float*)d_in[8];
  const float* Wo   = (const float*)d_in[9];
  const float* bo   = (const float*)d_in[10];
  const float* W1   = (const float*)d_in[11];
  const float* b1f  = (const float*)d_in[12];
  const float* W2   = (const float*)d_in[13];
  const float* b2f  = (const float*)d_in[14];
  const float* ln1g = (const float*)d_in[15];
  const float* ln1b = (const float*)d_in[16];
  const float* ln2g = (const float*)d_in[17];
  const float* ln2b = (const float*)d_in[18];
  (void)in_sizes; (void)n_in; (void)out_size; (void)ws_size;

  char* ws = (char*)d_ws;
  size_t off = 0;
  auto alloc = [&](size_t bytes) -> char* {
    char* p = ws + off;
    off += (bytes + 255) & ~(size_t)255;
    return p;
  };
  float* xf   = (float*)alloc((size_t)TOK * DM * 4);
  bf16*  xb   = (bf16*) alloc((size_t)TOK * DM * 2);
  float* mask = (float*)alloc((size_t)B_GRAPHS * NH * SEQ * SEQ * 4);
  bf16* wqkvb = (bf16*) alloc((size_t)DM * 3 * DM * 2);
  bf16* wob   = (bf16*) alloc((size_t)DM * DM * 2);
  bf16* w1b   = (bf16*) alloc((size_t)DM * DFF * 2);
  bf16* w2b   = (bf16*) alloc((size_t)DFF * DM * 2);
  bf16* qb    = (bf16*) alloc((size_t)TOK * DM * 2);
  bf16* kb    = (bf16*) alloc((size_t)TOK * DM * 2);
  bf16* vb    = (bf16*) alloc((size_t)TOK * DM * 2);
  bf16* ob    = (bf16*) alloc((size_t)TOK * DM * 2);
  bf16* hb    = (bf16*) alloc((size_t)TOK * DFF * 2);
  float* tmpf = (float*)alloc((size_t)TOK * DM * 4);

  const dim3 blk(256);

  // one-time preprocessing
  cvt_f32_bf16_kernel<<<(DM * 3 * DM + 255) / 256, blk, 0, stream>>>(Wqkv, wqkvb, DM * 3 * DM);
  cvt_f32_bf16_kernel<<<(DM * DM + 255) / 256,     blk, 0, stream>>>(Wo,   wob,   DM * DM);
  cvt_f32_bf16_kernel<<<(DM * DFF + 255) / 256,    blk, 0, stream>>>(W1,   w1b,   DM * DFF);
  cvt_f32_bf16_kernel<<<(DFF * DM + 255) / 256,    blk, 0, stream>>>(W2,   w2b,   DFF * DM);
  embed_kernel<<<(TOK * DM + 255) / 256, blk, 0, stream>>>(z, hgs, aemb, nemb, xf, xb);
  mask_kernel<<<(B_GRAPHS * SEQ * SEQ + 255) / 256, blk, 0, stream>>>(pos, demb, mask);

  // 8 shared-weight transformer layers
  for (int layer = 0; layer < 8; ++layer) {
    gemm_qkv_kernel<<<dim3(3 * DM / 128, TOK / 128), blk, 0, stream>>>(
        xb, wqkvb, bqkv, qb, kb, vb);
    attn_wmma_kernel<<<B_GRAPHS * NH, blk, 0, stream>>>(qb, kb, vb, mask, ob);
    gemm_wmma_kernel<<<dim3(DM / 128, TOK / 128), blk, 0, stream>>>(
        ob, wob, bo, tmpf, (bf16*)nullptr, TOK, DM, DM, 0);
    add_ln_kernel<<<TOK, blk, 0, stream>>>(xf, tmpf, ln1g, ln1b, xf, xb);
    gemm_wmma_kernel<<<dim3(DFF / 128, TOK / 128), blk, 0, stream>>>(
        xb, w1b, b1f, (float*)nullptr, hb, TOK, DFF, DM, 1);
    gemm_wmma_kernel<<<dim3(DM / 128, TOK / 128), blk, 0, stream>>>(
        hb, w2b, b2f, tmpf, (bf16*)nullptr, TOK, DM, DFF, 0);
    add_ln_kernel<<<TOK, blk, 0, stream>>>(xf, tmpf, ln2g, ln2b, xf, xb);
  }

  cls_extract_kernel<<<(B_GRAPHS * DM + 255) / 256, blk, 0, stream>>>(xf, (float*)d_out);
}
